// MoeLinearLoraLayer_91096256348304
// MI455X (gfx1250) — matrix operand order repaired and checked by reading, not compiled, not run on hardware
//
#include <hip/hip_runtime.h>
#include <hip/hip_bf16.h>
#include <stdint.h>

// Problem constants (match reference)
#define BB 2
#define SS 4096
#define DD 4096
#define OO 4096
#define RR 256
#define EE 8
#define MT (BB * SS)   // 8192 tokens
#define SCALE 2.0f

typedef unsigned short u16;
typedef __attribute__((ext_vector_type(16))) __bf16 v16bf;
typedef __attribute__((ext_vector_type(8)))  __bf16 v8bf;
typedef __attribute__((ext_vector_type(8)))  float  v8f;
typedef __attribute__((ext_vector_type(4)))  unsigned int v4u;
typedef __attribute__((ext_vector_type(8)))  int v8i_;
typedef __attribute__((ext_vector_type(4)))  int v4i_;

// ---------------- helpers ----------------

__device__ __forceinline__ u16 f2bf(float f) {
    union { float f; unsigned int u; } v; v.f = f;
    unsigned int u = v.u;
    u += 0x7FFFu + ((u >> 16) & 1u);   // round-to-nearest-even
    return (u16)(u >> 16);
}

__device__ __forceinline__ v8f wmma_bf16(v16bf a, v16bf b, v8f c) {
    return __builtin_amdgcn_wmma_f32_16x16x32_bf16(false, a, false, b, (short)0, c, false, false);
}

// Build a v16bf from two 16-byte chunks (keeps all LDS/global accesses 16B-aligned).
__device__ __forceinline__ v16bf frag16(const u16* p0, const u16* p1) {
    v8bf lo = *(const v8bf*)p0;
    v8bf hi = *(const v8bf*)p1;
    return __builtin_shufflevector(lo, hi, 0,1,2,3,4,5,6,7,8,9,10,11,12,13,14,15);
}

// A-fragment (16x32 bf16, row-major [M,K]): two 16B loads at +0 and +16 halves.
__device__ __forceinline__ v16bf load_afrag(const u16* rowp, int k0, int koff) {
    return frag16(rowp + k0 + koff, rowp + k0 + koff + 16);
}

// B-fragment (32x16 bf16, [N,K] row-major): two 16B loads at +0 and +16 halves.
__device__ __forceinline__ v16bf load_bfrag(const u16* colp, int k0, int kb) {
    return frag16(colp + k0 + kb / 2, colp + k0 + kb / 2 + 16);
}

// ---------------- Tensor Data Mover: 2D tile -> LDS ----------------
// data_size = 2B. Pads LDS dest: every 16 DWORDs (64B = one 32-elem row) add 4
// DWORDs (16B)  => LDS row stride = 80B (40 u16), bank-conflict-free fragments.

__device__ __forceinline__ void tdm_load_2d(unsigned lds_addr, const void* gaddr,
                                            unsigned tensor_d0, unsigned tensor_d1,
                                            unsigned tile_d0, unsigned tile_d1,
                                            unsigned stride0) {
    unsigned long long ga = (unsigned long long)(uintptr_t)gaddr;
    v4u g0;
    g0.x = 1u;                                   // count=1 (valid user descriptor)
    g0.y = lds_addr;                             // LDS byte address
    g0.z = (unsigned)(ga & 0xFFFFFFFFu);         // global_addr[31:0]
    g0.w = (unsigned)((ga >> 32) & 0x01FFFFFFu)  // global_addr[56:32]
         | (2u << 30);                           // type = 2 ("image")
    v8i_ g1;
    g1[0] = (int)((1u << 16)     // data_size = 1 -> 2 bytes
                | (1u << 20)     // pad_enable
                | (3u << 22)     // pad_interval code 3 = 16 DWORDs (64B)
                | (3u << 25));   // pad_amount  code 3 = 4 DWORDs (16B)
    g1[1] = (int)((tensor_d0 & 0xFFFFu) << 16);                         // tensor_dim0[15:0]
    g1[2] = (int)((tensor_d0 >> 16) | ((tensor_d1 & 0xFFFFu) << 16));   // d0 hi / d1 lo
    g1[3] = (int)((tensor_d1 >> 16) | (tile_d0 << 16));                 // d1 hi / tile_dim0
    g1[4] = (int)tile_d1;                                               // tile_dim1 (tile_dim2=0)
    g1[5] = (int)stride0;                                               // tensor_dim0_stride lo32
    g1[6] = 0;
    g1[7] = 0;
    v4i_ gz = {0, 0, 0, 0};
#if __clang_major__ >= 23
    v8i_ gz8 = {0, 0, 0, 0, 0, 0, 0, 0};
    __builtin_amdgcn_tensor_load_to_lds(g0, g1, gz, gz, gz8, 0);
#else
    __builtin_amdgcn_tensor_load_to_lds(g0, g1, gz, gz, 0);
#endif
}

// ---------------- conversion kernels ----------------

__global__ void cvt_f32_bf16(const float* __restrict__ in, u16* __restrict__ out, long n) {
    long i = (long)blockIdx.x * blockDim.x + threadIdx.x;
    long stride = (long)gridDim.x * blockDim.x;
    for (; i < n; i += stride) out[i] = f2bf(in[i]);
}

// lora_A [E,D,R] f32 -> At [E,R,D] bf16
__global__ void cvt_at(const float* __restrict__ in, u16* __restrict__ out) {
    long n = (long)EE * RR * DD;
    long i = (long)blockIdx.x * blockDim.x + threadIdx.x;
    long stride = (long)gridDim.x * blockDim.x;
    for (; i < n; i += stride) {
        long e = i / ((long)RR * DD);
        long rem = i - e * (long)RR * DD;
        long r = rem / DD;
        long d = rem - r * DD;
        out[i] = f2bf(in[e * (long)DD * RR + d * RR + r]);
    }
}

// lora_B [E,R,O] f32 -> Bt [E,O,R] bf16
__global__ void cvt_bt(const float* __restrict__ in, u16* __restrict__ out) {
    long n = (long)EE * OO * RR;
    long i = (long)blockIdx.x * blockDim.x + threadIdx.x;
    long stride = (long)gridDim.x * blockDim.x;
    for (; i < n; i += stride) {
        long e = i / ((long)OO * RR);
        long rem = i - e * (long)OO * RR;
        long o = rem / RR;
        long r = rem - o * RR;
        out[i] = f2bf(in[e * (long)RR * OO + r * OO + o]);
    }
}

// ---------------- expert token lists ----------------

__global__ void build_lists(const int* __restrict__ tt, int* __restrict__ counts,
                            int* __restrict__ lists) {
    int e = blockIdx.x;
    __shared__ int cnt;
    if (threadIdx.x == 0) cnt = 0;
    __syncthreads();
    for (int s = threadIdx.x; s < SS; s += blockDim.x) {
        if (tt[s] == e) {
            int p = atomicAdd(&cnt, 1);
            lists[e * SS + p] = s;
        }
    }
    __syncthreads();
    if (threadIdx.x == 0) counts[e] = cnt;
}

// ---------------- base GEMM: out[MT,O] = Xb[MT,D] * Wb[O,D]^T ----------------
// Workgroup (8 waves) computes 128x128 C tile. Waves 0/1 drive the Tensor Data
// Mover (X-tile / W-tile DMAs, double-buffered 128x32 bf16, TENSORcnt), all 8
// waves compute 32x64 sub-tiles from LDS fragments. K-loop is manually
// unrolled x2 so LDS buffer pointers are compile-time constants (keeps WMMA
// accumulators in place, no PHI copies).

#define TM 128
#define TN 128
#define KT 32
#define LROW 40   // u16 per LDS row (32 data + 8 pad) = 80 bytes

__device__ __forceinline__ void compute_step(const u16* __restrict__ lXb,
                                             const u16* __restrict__ lWb,
                                             int msub, int nsub, int mlo,
                                             int koff, int kb, v8f acc[8]) {
    const u16* xr0 = lXb + (msub + mlo) * LROW;
    const u16* xr1 = xr0 + 16 * LROW;
    v16bf af0 = frag16(xr0 + koff, xr0 + koff + 16);
    v16bf af1 = frag16(xr1 + koff, xr1 + koff + 16);
    #pragma unroll
    for (int nn = 0; nn < 4; ++nn) {
        const u16* wp = lWb + (nsub + nn * 16 + mlo) * LROW + kb / 2;
        v16bf bf = frag16(wp, wp + 16);
        acc[nn]     = wmma_bf16(af0, bf, acc[nn]);
        acc[4 + nn] = wmma_bf16(af1, bf, acc[4 + nn]);
    }
}

__global__ __launch_bounds__(256) void gemm_base(const u16* __restrict__ Xb,
                                                 const u16* __restrict__ Wb,
                                                 float* __restrict__ out) {
    __shared__ u16 lX[2][TM * LROW];
    __shared__ u16 lW[2][TN * LROW];

    int tmb = blockIdx.x * TM;
    int tnb = blockIdx.y * TN;
    int tid = threadIdx.x;
    int lane = tid & 31;
    int wid = __builtin_amdgcn_readfirstlane(tid >> 5);   // wave-uniform SGPR
    int msub = (wid & 3) * 32;
    int nsub = (wid >> 2) * 64;
    int mlo = lane & 15;
    int kh  = lane >> 4;
    int koff = kh * 8;
    int kb   = kh * 16;

    const u16* xg = Xb + (size_t)tmb * DD;
    const u16* wg = Wb + (size_t)tnb * DD;

    // preload buffer 0 (wave0: X tile, wave1: W tile)
    if (wid == 0) tdm_load_2d((unsigned)(uintptr_t)&lX[0][0], xg, DD, MT, KT, TM, DD);
    if (wid == 1) tdm_load_2d((unsigned)(uintptr_t)&lW[0][0], wg, DD, OO, KT, TN, DD);
    __builtin_amdgcn_s_wait_tensorcnt((short)0);   // no-op for non-issuing waves
    __syncthreads();

    v8f zero = {0.f,0.f,0.f,0.f,0.f,0.f,0.f,0.f};
    v8f acc[8];
    #pragma unroll
    for (int q = 0; q < 8; ++q) acc[q] = zero;

    const int KSTEPS = DD / KT;                    // 128, even
    for (int kt = 0; kt < KSTEPS; kt += 2) {
        // ---- half A: compute buf0, prefetch (kt+1) into buf1 ----
        {
            int k0n = (kt + 1) * KT;
            if (wid == 0) tdm_load_2d((unsigned)(uintptr_t)&lX[1][0], xg + k0n, DD, MT, KT, TM, DD);
            if (wid == 1) tdm_load_2d((unsigned)(uintptr_t)&lW[1][0], wg + k0n, DD, OO, KT, TN, DD);
        }
        compute_step(&lX[0][0], &lW[0][0], msub, nsub, mlo, koff, kb, acc);
        __builtin_amdgcn_s_wait_tensorcnt((short)0);
        __syncthreads();

        // ---- half B: compute buf1, prefetch (kt+2) into buf0 ----
        if (kt + 2 < KSTEPS) {
            int k0n = (kt + 2) * KT;
            if (wid == 0) tdm_load_2d((unsigned)(uintptr_t)&lX[0][0], xg + k0n, DD, MT, KT, TM, DD);
            if (wid == 1) tdm_load_2d((unsigned)(uintptr_t)&lW[0][0], wg + k0n, DD, OO, KT, TN, DD);
        }
        compute_step(&lX[1][0], &lW[1][0], msub, nsub, mlo, koff, kb, acc);
        __builtin_amdgcn_s_wait_tensorcnt((short)0);
        __syncthreads();
    }

    int colbase = tnb + nsub + mlo;
    #pragma unroll
    for (int half = 0; half < 2; ++half) {
        int rowbase = tmb + msub + half * 16 + kh * 8;
        #pragma unroll
        for (int nn = 0; nn < 4; ++nn) {
            #pragma unroll
            for (int j = 0; j < 8; ++j) {
                out[((size_t)(rowbase + j)) * OO + colbase + nn * 16] = acc[half * 4 + nn][j];
            }
        }
    }
}

// ---------------- LoRA GEMM1: h = gather(x)·A_e  (K=D, N=R), store bf16 ----------------

__global__ __launch_bounds__(32) void gemm_h(const u16* __restrict__ Xb,
                                             const u16* __restrict__ At,
                                             const int* __restrict__ lists,
                                             const int* __restrict__ counts,
                                             u16* __restrict__ Hb) {
    int i  = blockIdx.x;
    int e  = blockIdx.y;
    int z  = blockIdx.z;
    int b  = z >> 2;          // R/64 = 4 col tiles
    int nt = z & 3;
    int count = counts[e];
    int i16 = i * 16;
    if (i16 >= count) return;

    __shared__ int ss[16];
    int lane = threadIdx.x;
    if (lane < 16) {
        int g = i16 + lane;
        ss[lane] = lists[e * SS + (g < count ? g : count - 1)];
    }
    __syncthreads();

    int mlo = lane & 15;
    int kh  = lane >> 4;
    int koff = kh * 8;
    int kb   = kh * 16;
    int n0 = nt * 64;

    const u16* arow = Xb + ((size_t)(b * SS + ss[mlo])) * DD;
    const u16* bbase = At + (size_t)e * RR * DD + ((size_t)(n0 + mlo)) * DD;

    v8f zero = {0.f,0.f,0.f,0.f,0.f,0.f,0.f,0.f};
    v8f acc[4];
    #pragma unroll
    for (int q = 0; q < 4; ++q) acc[q] = zero;

    for (int k0 = 0; k0 < DD; k0 += 32) {
        v16bf af = load_afrag(arow, k0, koff);
        #pragma unroll
        for (int nn = 0; nn < 4; ++nn) {
            v16bf bf = load_bfrag(bbase + (size_t)nn * 16 * DD, k0, kb);
            acc[nn] = wmma_bf16(af, bf, acc[nn]);
        }
    }

    #pragma unroll
    for (int nn = 0; nn < 4; ++nn) {
        #pragma unroll
        for (int j = 0; j < 8; ++j) {
            int m = j + kh * 8;
            if (i16 + m < count) {
                int s = ss[m];
                Hb[((size_t)(b * SS + s)) * RR + n0 + nn * 16 + mlo] = f2bf(acc[nn][j]);
            }
        }
    }
}

// ---------------- LoRA GEMM2: out += SCALE * gather(h)·B_e  (K=R, N=O) ----------------

__global__ __launch_bounds__(32) void gemm_o(const u16* __restrict__ Hb,
                                             const u16* __restrict__ Bt,
                                             const int* __restrict__ lists,
                                             const int* __restrict__ counts,
                                             float* __restrict__ out) {
    int i  = blockIdx.x;
    int e  = blockIdx.y;
    int z  = blockIdx.z;
    int b  = z >> 6;          // O/64 = 64 col tiles
    int nt = z & 63;
    int count = counts[e];
    int i16 = i * 16;
    if (i16 >= count) return;

    __shared__ int ss[16];
    int lane = threadIdx.x;
    if (lane < 16) {
        int g = i16 + lane;
        ss[lane] = lists[e * SS + (g < count ? g : count - 1)];
    }
    __syncthreads();

    int mlo = lane & 15;
    int kh  = lane >> 4;
    int koff = kh * 8;
    int kb   = kh * 16;
    int n0 = nt * 64;

    const u16* arow = Hb + ((size_t)(b * SS + ss[mlo])) * RR;
    const u16* bbase = Bt + (size_t)e * OO * RR + ((size_t)(n0 + mlo)) * RR;

    v8f zero = {0.f,0.f,0.f,0.f,0.f,0.f,0.f,0.f};
    v8f acc[4];
    #pragma unroll
    for (int q = 0; q < 4; ++q) acc[q] = zero;

    #pragma unroll
    for (int k0 = 0; k0 < RR; k0 += 32) {
        v16bf af = load_afrag(arow, k0, koff);
        #pragma unroll
        for (int nn = 0; nn < 4; ++nn) {
            v16bf bf = load_bfrag(bbase + (size_t)nn * 16 * RR, k0, kb);
            acc[nn] = wmma_bf16(af, bf, acc[nn]);
        }
    }

    #pragma unroll
    for (int nn = 0; nn < 4; ++nn) {
        #pragma unroll
        for (int j = 0; j < 8; ++j) {
            int m = j + kh * 8;
            if (i16 + m < count) {
                int s = ss[m];
                size_t idx = ((size_t)(b * SS + s)) * OO + n0 + nn * 16 + mlo;
                out[idx] += SCALE * acc[nn][j];   // disjoint rows per expert: no atomics needed
            }
        }
    }
}

// ---------------- launch ----------------

extern "C" void kernel_launch(void* const* d_in, const int* in_sizes, int n_in,
                              void* d_out, int out_size, void* d_ws, size_t ws_size,
                              hipStream_t stream) {
    const float* x  = (const float*)d_in[0];   // [B,S,D]
    const float* w  = (const float*)d_in[1];   // [O,D]
    const float* la = (const float*)d_in[2];   // [E,D,R]
    const float* lb = (const float*)d_in[3];   // [E,R,O]
    const int*   tt = (const int*)d_in[4];     // [S]
    float* out = (float*)d_out;                // [B,S,O]

    char* ws = (char*)d_ws;
    size_t o_xb  = 0;
    size_t o_wb  = o_xb  + (size_t)MT * DD * 2;       // x bf16
    size_t o_at  = o_wb  + (size_t)OO * DD * 2;       // W bf16
    size_t o_bt  = o_at  + (size_t)EE * RR * DD * 2;  // A^T bf16
    size_t o_hb  = o_bt  + (size_t)EE * OO * RR * 2;  // B^T bf16
    size_t o_cnt = o_hb  + (size_t)MT * RR * 2;       // h bf16
    size_t o_lst = o_cnt + 256;

    u16* xb = (u16*)(ws + o_xb);
    u16* wb = (u16*)(ws + o_wb);
    u16* at = (u16*)(ws + o_at);
    u16* bt = (u16*)(ws + o_bt);
    u16* hb = (u16*)(ws + o_hb);
    int* cnt = (int*)(ws + o_cnt);
    int* lst = (int*)(ws + o_lst);

    cvt_f32_bf16<<<2048, 256, 0, stream>>>(x, xb, (long)MT * DD);
    cvt_f32_bf16<<<2048, 256, 0, stream>>>(w, wb, (long)OO * DD);
    cvt_at<<<1024, 256, 0, stream>>>(la, at);
    cvt_bt<<<1024, 256, 0, stream>>>(lb, bt);
    build_lists<<<EE, 256, 0, stream>>>(tt, cnt, lst);

    gemm_base<<<dim3(MT / TM, OO / TN), 256, 0, stream>>>(xb, wb, out);
    gemm_h<<<dim3(SS / 16, EE, BB * (RR / 64)), 32, 0, stream>>>(xb, at, lst, cnt, hb);
    gemm_o<<<dim3(SS / 16, EE, BB * (OO / 64)), 32, 0, stream>>>(hb, bt, lst, cnt, out);
}
